// Transformer_58712202936482
// MI455X (gfx1250) — compile-verified, matrix operand order checked by reading, not run on hardware
//
#include <hip/hip_runtime.h>
#include <math.h>

// ---------------------------------------------------------------------------
// Transformer forward B=2,S=1024,V=512,D=1024,H=16,HS=64,L=4,DFF=4096.
// bf16-resident dataflow: weights are transposed+converted to bf16 once per
// layer, activations carry bf16 shadows. GEMM: v_wmma_f32_16x16x32_bf16 with
// triple-buffered GLOBAL_LOAD_ASYNC_TO_LDS_B128 staging. Attention: fused
// flash kernel (online softmax, permlane16 row reductions).
// ---------------------------------------------------------------------------

typedef __bf16 bf16_t;
typedef __attribute__((ext_vector_type(16))) __bf16 v16bf;
typedef __attribute__((ext_vector_type(8)))  __bf16 v8bf;
typedef __attribute__((ext_vector_type(8)))  float  v8f;
typedef __attribute__((ext_vector_type(4)))  int    v4i;

#define WMMA_BF16(a, b, c) \
  __builtin_amdgcn_wmma_f32_16x16x32_bf16(false, (a), false, (b), (short)0, (c), false, false)

// ---- async global->LDS copy (CDNA5), guarded with sync fallback -----------
#if __has_builtin(__builtin_amdgcn_global_load_async_to_lds_b128)
#define HAVE_ASYNC 1
typedef __attribute__((address_space(1))) v4i* as1_v4i;
typedef __attribute__((address_space(3))) v4i* as3_v4i;
__device__ __forceinline__ void async_cp16(const bf16_t* g, bf16_t* l) {
  __builtin_amdgcn_global_load_async_to_lds_b128(
      (as1_v4i)(void*)g, (as3_v4i)(void*)l, 0, 0);
}
#else
#define HAVE_ASYNC 0
__device__ __forceinline__ void async_cp16(const bf16_t* g, bf16_t* l) {
  *(v8bf*)l = *(const v8bf*)g;   // sync fallback
}
#endif

__device__ __forceinline__ void wait_async_0() {
#if HAVE_ASYNC
#if __has_builtin(__builtin_amdgcn_s_wait_asynccnt)
  __builtin_amdgcn_s_wait_asynccnt(0);
#else
  asm volatile("s_wait_asynccnt 0x0" ::: "memory");
#endif
#endif
}
__device__ __forceinline__ void wait_async_3() {
#if HAVE_ASYNC
#if __has_builtin(__builtin_amdgcn_s_wait_asynccnt)
  __builtin_amdgcn_s_wait_asynccnt(3);
#else
  asm volatile("s_wait_asynccnt 0x3" ::: "memory");
#endif
#endif
}

// ---- 16-lane-row XOR shuffle via v_permlane16_b32 (VALU, no LDS pipe) -----
#if __has_builtin(__builtin_amdgcn_permlane16)
__device__ __forceinline__ float lanexor(float x, int step) {  // step->xor(1<<step)
  const unsigned lo[4] = {0x67452301u, 0x54761032u, 0x32107654u, 0xFEDCBA98u};
  const unsigned hi[4] = {0xEFCDAB89u, 0xDCFE98BAu, 0xBA98FEDCu, 0x76543210u};
  unsigned r = __builtin_amdgcn_permlane16(__float_as_uint(x), __float_as_uint(x),
                                           lo[step], hi[step], false, false);
  return __uint_as_float(r);
}
#else
__device__ __forceinline__ float lanexor(float x, int step) {
  return __shfl_xor(x, 1 << step, 32);
}
#endif

// CDNA5 16-bit A/B fragment (ISA 7.12.2): lane = M (A) / N (B) index, lane
// half hl picks K {hl*8..+7, 16+hl*8..+7} inside a 32-K block. Row-contiguous
// data -> two 16B loads. Works for LDS and global pointers.
__device__ __forceinline__ v16bf frag_ld(const bf16_t* row, int hl, int kblk) {
  const bf16_t* p = row + kblk * 32 + hl * 8;
  v8bf lo = *(const v8bf*)(p);
  v8bf hi = *(const v8bf*)(p + 16);
  v16bf r;
#pragma unroll
  for (int i = 0; i < 8; ++i) { r[i] = lo[i]; r[i + 8] = hi[i]; }
  return r;
}

// ---------------------------------------------------------------------------
// GEMM: out = epilogue(A[MxK]bf16 @ Bt[NxK]^T bf16 + bias [+resid f32])
// ACT: 0 none, 1 exact GELU.  OUT: bit0 f32 out, bit1 bf16 out.
// 256 thr = 8 waves (4x2), wave tile 32x32, BK=32, triple-buffered async LDS.
// ---------------------------------------------------------------------------
template <int ACT, bool RESID, int OUT>
__global__ __launch_bounds__(256) void gemm_bf16(
    const bf16_t* __restrict__ A, const bf16_t* __restrict__ Bt,
    const float* __restrict__ bias, const float* __restrict__ resid,
    float* __restrict__ outF, bf16_t* __restrict__ outB,
    int M, int N, int K) {
  constexpr int BM = 128, BN = 64, BK = 32, LDT = 48;  // 96B rows, 16B aligned
  __shared__ bf16_t As[3][BM * LDT];
  __shared__ bf16_t Bs[3][BN * LDT];

  const int tid = threadIdx.x;
  const int lane = tid & 31, wave = tid >> 5;
  const int hl = lane >> 4, ln = lane & 15;
  const int wm = wave >> 1, wn = wave & 1;
  const int m0 = blockIdx.y * BM, n0 = blockIdx.x * BN;
  const int NT = K / BK;

  auto issue = [&](int t) {  // 3 async b128 per thread per tile
    const int sl = t % 3, k0 = t * BK;
#pragma unroll
    for (int i = 0; i < 2; ++i) {               // A tile: 128x32 bf16 = 8KB
      int cc = tid * 2 + i;
      int row = cc >> 2, c16 = (cc & 3) * 8;
      async_cp16(A + (size_t)(m0 + row) * K + k0 + c16, &As[sl][row * LDT + c16]);
    }
    {                                           // B tile: 64x32 bf16 = 4KB
      int row = tid >> 2, c16 = (tid & 3) * 8;
      async_cp16(Bt + (size_t)(n0 + row) * K + k0 + c16, &Bs[sl][row * LDT + c16]);
    }
  };

  v8f acc[2][2] = {};
  issue(0);
  if (NT > 1) issue(1);

  for (int t = 0; t < NT; ++t) {
    if (t + 1 < NT) wait_async_3(); else wait_async_0();  // tile t landed
    __syncthreads();                                      // ... in every wave
    if (t + 2 < NT) issue(t + 2);  // safe: slot (t+2)%3 readers done pre-barrier
    const bf16_t* as = As[t % 3];
    const bf16_t* bs = Bs[t % 3];
    v16bf a0 = frag_ld(as + (wm * 32 + ln) * LDT, hl, 0);
    v16bf a1 = frag_ld(as + (wm * 32 + 16 + ln) * LDT, hl, 0);
    v16bf b0 = frag_ld(bs + (wn * 32 + ln) * LDT, hl, 0);
    v16bf b1 = frag_ld(bs + (wn * 32 + 16 + ln) * LDT, hl, 0);
    acc[0][0] = WMMA_BF16(a0, b0, acc[0][0]);
    acc[0][1] = WMMA_BF16(a0, b1, acc[0][1]);
    acc[1][0] = WMMA_BF16(a1, b0, acc[1][0]);
    acc[1][1] = WMMA_BF16(a1, b1, acc[1][1]);
  }

  // epilogue: C/D layout (M = r + 8*hl, N = ln)
#pragma unroll
  for (int i = 0; i < 2; ++i)
#pragma unroll
    for (int j = 0; j < 2; ++j)
#pragma unroll
      for (int r = 0; r < 8; ++r) {
        int row = m0 + wm * 32 + i * 16 + r + 8 * hl;
        int col = n0 + wn * 32 + j * 16 + ln;
        float v = acc[i][j][r] + bias[col];
        if (ACT == 1) v = 0.5f * v * (1.0f + erff(v * 0.70710678118654752f));
        if (RESID) v += resid[(size_t)row * N + col];
        if (OUT & 1) outF[(size_t)row * N + col] = v;
        if (OUT & 2) outB[(size_t)row * N + col] = (bf16_t)v;
      }
}

// ---------------------------------------------------------------------------
// LayerNorm per row (D=1024), dual f32 + bf16 outputs
// ---------------------------------------------------------------------------
__global__ __launch_bounds__(256) void ln_kernel(
    const float* __restrict__ x, const float* __restrict__ g,
    const float* __restrict__ b, float* __restrict__ h,
    bf16_t* __restrict__ hb, int D) {
  __shared__ float red[256];
  int row = blockIdx.x, tid = threadIdx.x;
  const float* xr = x + (size_t)row * D;
  float s = 0.f;
  for (int i = tid; i < D; i += 256) s += xr[i];
  red[tid] = s; __syncthreads();
  for (int o = 128; o > 0; o >>= 1) { if (tid < o) red[tid] += red[tid + o]; __syncthreads(); }
  float mean = red[0] / D; __syncthreads();
  float vs = 0.f;
  for (int i = tid; i < D; i += 256) { float d = xr[i] - mean; vs += d * d; }
  red[tid] = vs; __syncthreads();
  for (int o = 128; o > 0; o >>= 1) { if (tid < o) red[tid] += red[tid + o]; __syncthreads(); }
  float rstd = rsqrtf(red[0] / D + 1e-5f);
  float* hr = h + (size_t)row * D;
  bf16_t* hbr = hb + (size_t)row * D;
  for (int i = tid; i < D; i += 256) {
    float v = (xr[i] - mean) * rstd * g[i] + b[i];
    hr[i] = v; hbr[i] = (bf16_t)v;
  }
}

// ---------------------------------------------------------------------------
// Weight prep: transpose + f32->bf16.  Bt[n*K+k] = W[k*N+n]
// ---------------------------------------------------------------------------
__global__ __launch_bounds__(256) void transpose_cvt(
    const float* __restrict__ W, bf16_t* __restrict__ Bt, int K, int N) {
  int idx = blockIdx.x * 256 + threadIdx.x;
  if (idx >= K * N) return;
  int k = idx % K, n = idx / K;  // k fastest -> coalesced writes
  Bt[(size_t)n * K + k] = (bf16_t)W[(size_t)k * N + n];
}

// Wqkv[l] (H, D, 3HS) -> Bt[(hh*192+e) * D + d]
__global__ __launch_bounds__(256) void transpose_wqkv(
    const float* __restrict__ Wl, bf16_t* __restrict__ Bt) {
  int idx = blockIdx.x * 256 + threadIdx.x;
  if (idx >= 3072 * 1024) return;
  int d = idx & 1023, n = idx >> 10;
  int e = n % 192, hh = n / 192;
  Bt[(size_t)n * 1024 + d] = (bf16_t)Wl[((size_t)hh * 1024 + d) * 192 + e];
}

__global__ __launch_bounds__(256) void cvt_bf16(
    const float* __restrict__ x, bf16_t* __restrict__ y, int n) {
  int i = blockIdx.x * 256 + threadIdx.x;
  if (i < n) y[i] = (bf16_t)x[i];
}

// ---------------------------------------------------------------------------
// Flash attention, bf16 QKV in / bf16 O out.  Block = (b, h, 64 Q rows),
// 4 waves; 64-key tiles; WMMA for QK^T and P@V; online softmax.
// QKV element (b,s,h,c): [((b*S+s)*H+h)*192 + c],  q:c<64  k:64..127 v:128..191
// ---------------------------------------------------------------------------
__global__ __launch_bounds__(128) void flash_attn(
    const bf16_t* __restrict__ QKV, bf16_t* __restrict__ O) {
  constexpr int LKV = 72;                // 144B rows (16B aligned)
  __shared__ bf16_t Ks[64 * LKV];        // Ks[key][e]
  __shared__ bf16_t Vs[64 * LKV];        // Vs[e][key] (transposed)
  __shared__ bf16_t Ps[4][16 * LKV];     // per-wave P tile

  const int b = blockIdx.z, h = blockIdx.y;
  const int q0 = blockIdx.x * 64;
  const int tid = threadIdx.x, lane = tid & 31, wave = tid >> 5;
  const int hl = lane >> 4, ln = lane & 15;
  const float scale = 0.125f;  // 1/sqrt(64), applied to scores

  auto qkv_at = [&](int s, int off) -> const bf16_t* {
    return QKV + ((size_t)(b * 1024 + s) * 16 + h) * 192 + off;
  };

  // Q fragments: global reads already match the A-fragment pattern
  const bf16_t* qrow = qkv_at(q0 + wave * 16 + ln, 0);
  v16bf qf0 = frag_ld(qrow, hl, 0);
  v16bf qf1 = frag_ld(qrow, hl, 1);

  v8f oacc[4] = {};
  float mrow[8], lrow[8];
#pragma unroll
  for (int r = 0; r < 8; ++r) { mrow[r] = -1e30f; lrow[r] = 0.f; }

  for (int t0 = 0; t0 < 1024; t0 += 64) {
    // K tile: raw async copies (row-major, padded rows)
#pragma unroll
    for (int i = 0; i < 4; ++i) {
      int cc = tid * 4 + i;
      int kv = cc >> 3, c8 = (cc & 7) * 8;
      async_cp16(qkv_at(t0 + kv, 64) + c8, Ks + kv * LKV + c8);
    }
    // V tile transposed: pair-packed b32 LDS stores (overlaps with K asyncs)
    {
      int vp = tid >> 3, e8 = (tid & 7) * 8;
#pragma unroll
      for (int i = 0; i < 2; ++i) {
        int k0 = (vp + i * 16) * 2;
        v8bf va = *(const v8bf*)(qkv_at(t0 + k0, 128) + e8);
        v8bf vb = *(const v8bf*)(qkv_at(t0 + k0 + 1, 128) + e8);
#pragma unroll
        for (int c = 0; c < 8; ++c) {
          unsigned pk = (unsigned)__builtin_bit_cast(unsigned short, va[c]) |
                        ((unsigned)__builtin_bit_cast(unsigned short, vb[c]) << 16);
          *(unsigned*)(Vs + (e8 + c) * LKV + k0) = pk;
        }
      }
    }
    wait_async_0();
    __syncthreads();

    // S = Q @ K^T (4 subtiles of 16 keys), then scale
    v8f sc[4];
#pragma unroll
    for (int j = 0; j < 4; ++j) {
      const bf16_t* kr = Ks + (j * 16 + ln) * LKV;
      v8f c = {};
      c = WMMA_BF16(qf0, frag_ld(kr, hl, 0), c);
      c = WMMA_BF16(qf1, frag_ld(kr, hl, 1), c);
#pragma unroll
      for (int r = 0; r < 8; ++r) c[r] *= scale;
      sc[j] = c;
    }

    // online softmax (row r lives in one 16-lane row: M = r + 8*hl)
#pragma unroll
    for (int r = 0; r < 8; ++r) {
      float mx = fmaxf(fmaxf(sc[0][r], sc[1][r]), fmaxf(sc[2][r], sc[3][r]));
#pragma unroll
      for (int stp = 0; stp < 4; ++stp) mx = fmaxf(mx, lanexor(mx, stp));
      float mnew = fmaxf(mrow[r], mx);
      float corr = __expf(mrow[r] - mnew);
      mrow[r] = mnew;
      float rs = 0.f;
#pragma unroll
      for (int j = 0; j < 4; ++j) {
        float p = __expf(sc[j][r] - mnew);
        sc[j][r] = p;
        rs += p;
      }
#pragma unroll
      for (int stp = 0; stp < 4; ++stp) rs += lanexor(rs, stp);
      lrow[r] = lrow[r] * corr + rs;
#pragma unroll
      for (int j = 0; j < 4; ++j) oacc[j][r] *= corr;
    }

    // P (C-layout) -> per-wave LDS -> A-layout fragments
#pragma unroll
    for (int r = 0; r < 8; ++r) {
      int prow = r + 8 * hl;
#pragma unroll
      for (int j = 0; j < 4; ++j)
        Ps[wave][prow * LKV + j * 16 + ln] = (bf16_t)sc[j][r];
    }
    asm volatile("s_wait_dscnt 0x0" ::: "memory");  // same-wave LDS RAW

    v16bf pf0 = frag_ld(Ps[wave] + ln * LKV, hl, 0);
    v16bf pf1 = frag_ld(Ps[wave] + ln * LKV, hl, 1);
#pragma unroll
    for (int j = 0; j < 4; ++j) {
      const bf16_t* vr = Vs + (j * 16 + ln) * LKV;
      oacc[j] = WMMA_BF16(pf0, frag_ld(vr, hl, 0), oacc[j]);
      oacc[j] = WMMA_BF16(pf1, frag_ld(vr, hl, 1), oacc[j]);
    }
    __syncthreads();
  }

  // O[(b*S+s)*D + h*64 + e] (bf16, feeds Wo GEMM)
#pragma unroll
  for (int r = 0; r < 8; ++r) {
    int s = q0 + wave * 16 + r + 8 * hl;
    float inv = 1.0f / lrow[r];
#pragma unroll
    for (int j = 0; j < 4; ++j)
      O[(size_t)(b * 1024 + s) * 1024 + h * 64 + j * 16 + ln] =
          (bf16_t)(oacc[j][r] * inv);
  }
}

// ---------------------------------------------------------------------------
// Row softmax over V=512 (final output, f32)
// ---------------------------------------------------------------------------
__global__ __launch_bounds__(128) void softmax_rows(
    const float* __restrict__ logits, float* __restrict__ out, int N) {
  __shared__ float red[128];
  int row = blockIdx.x, tid = threadIdx.x;
  const float* xr = logits + (size_t)row * N;
  float mx = -1e30f;
  for (int i = tid; i < N; i += 128) mx = fmaxf(mx, xr[i]);
  red[tid] = mx; __syncthreads();
  for (int o = 64; o > 0; o >>= 1) { if (tid < o) red[tid] = fmaxf(red[tid], red[tid + o]); __syncthreads(); }
  mx = red[0]; __syncthreads();
  float s = 0.f;
  for (int i = tid; i < N; i += 128) s += __expf(xr[i] - mx);
  red[tid] = s; __syncthreads();
  for (int o = 64; o > 0; o >>= 1) { if (tid < o) red[tid] += red[tid + o]; __syncthreads(); }
  float inv = 1.f / red[0];
  float* orow = out + (size_t)row * N;
  for (int i = tid; i < N; i += 128) orow[i] = __expf(xr[i] - mx) * inv;
}

// ---------------------------------------------------------------------------
extern "C" void kernel_launch(void* const* d_in, const int* in_sizes, int n_in,
                              void* d_out, int out_size, void* d_ws, size_t ws_size,
                              hipStream_t stream) {
  const float* tokens = (const float*)d_in[0];
  const float* W_in   = (const float*)d_in[1];
  const float* b_in   = (const float*)d_in[2];
  const float* ln_g   = (const float*)d_in[3];
  const float* ln_b   = (const float*)d_in[4];
  const float* Wqkv   = (const float*)d_in[5];
  const float* bqkv   = (const float*)d_in[6];
  const float* Wo     = (const float*)d_in[7];
  const float* bo     = (const float*)d_in[8];
  const float* W1     = (const float*)d_in[9];
  const float* b1     = (const float*)d_in[10];
  const float* W2     = (const float*)d_in[11];
  const float* b2     = (const float*)d_in[12];
  const float* W_out  = (const float*)d_in[13];
  const float* b_out  = (const float*)d_in[14];
  float* out = (float*)d_out;

  constexpr int Bc = 2, S = 1024, V = 512, D = 1024, H = 16, HS = 64, L = 4,
                DFF = 4096, M = Bc * S;

  // ---- workspace carve (f32 region then bf16 region), ~100 MB total ----
  float* wsf  = (float*)d_ws;
  float* bufX = wsf;                          // M*D
  float* bufH = bufX + (size_t)M * D;         // M*D
  float* bufT = bufH + (size_t)M * D;         // M*D
  float* bufL = bufT + (size_t)M * D;         // M*V
  bf16_t* wsb    = (bf16_t*)(bufL + (size_t)M * V);
  bf16_t* tokB   = wsb;                             // M*V
  bf16_t* bufHb  = tokB   + (size_t)M * V;          // M*D
  bf16_t* bufQKb = bufHb  + (size_t)M * D;          // M*3D
  bf16_t* bufOb  = bufQKb + (size_t)M * 3 * D;      // M*D
  bf16_t* bufMhb = bufOb  + (size_t)M * D;          // M*DFF
  bf16_t* bufXb  = bufMhb + (size_t)M * DFF;        // M*D
  bf16_t* WinT   = bufXb  + (size_t)M * D;          // D*V
  bf16_t* WoutT  = WinT   + (size_t)D * V;          // V*D
  bf16_t* WqkvT  = WoutT  + (size_t)V * D;          // 3D*D
  bf16_t* WoT    = WqkvT  + (size_t)3 * D * D;      // D*D
  bf16_t* W1T    = WoT    + (size_t)D * D;          // DFF*D
  bf16_t* W2T    = W1T    + (size_t)DFF * D;        // D*DFF

  dim3 blk(256);

  // one-time prep
  cvt_bf16<<<(M * V) / 256, 256, 0, stream>>>(tokens, tokB, M * V);
  transpose_cvt<<<(V * D) / 256, 256, 0, stream>>>(W_in, WinT, V, D);
  transpose_cvt<<<(D * V) / 256, 256, 0, stream>>>(W_out, WoutT, D, V);

  // x = tokens @ W_in + b_in
  gemm_bf16<0, false, 1><<<dim3(D / 64, M / 128), blk, 0, stream>>>(
      tokB, WinT, b_in, nullptr, bufX, nullptr, M, D, V);

  for (int l = 0; l < L; ++l) {
    ln_kernel<<<M, 256, 0, stream>>>(bufX, ln_g + (size_t)l * D,
                                     ln_b + (size_t)l * D, bufH, bufHb, D);
    transpose_wqkv<<<(3 * D * D) / 256, 256, 0, stream>>>(
        Wqkv + (size_t)l * H * D * 3 * HS, WqkvT);
    transpose_cvt<<<(D * D) / 256, 256, 0, stream>>>(
        Wo + (size_t)l * D * D, WoT, D, D);
    transpose_cvt<<<(D * DFF) / 256, 256, 0, stream>>>(
        W1 + (size_t)l * D * DFF, W1T, D, DFF);
    transpose_cvt<<<(DFF * D) / 256, 256, 0, stream>>>(
        W2 + (size_t)l * DFF * D, W2T, DFF, D);

    // qkv = h @ Wqkv + bqkv   (bf16 out only -> flash input)
    gemm_bf16<0, false, 2><<<dim3(3 * D / 64, M / 128), blk, 0, stream>>>(
        bufHb, WqkvT, bqkv + (size_t)l * H * 3 * HS, nullptr, nullptr, bufQKb,
        M, 3 * D, D);
    flash_attn<<<dim3(S / 64, H, Bc), 128, 0, stream>>>(bufQKb, bufOb);
    // t = o @ Wo + bo + h   (f32, residual source only)
    gemm_bf16<0, true, 1><<<dim3(D / 64, M / 128), blk, 0, stream>>>(
        bufOb, WoT, bo + (size_t)l * D, bufH, bufT, nullptr, M, D, D);
    // m = gelu(h @ W1 + b1)  (bf16 out only -> W2 GEMM input)
    gemm_bf16<1, false, 2><<<dim3(DFF / 64, M / 128), blk, 0, stream>>>(
        bufHb, W1T, b1 + (size_t)l * DFF, nullptr, nullptr, bufMhb, M, DFF, D);
    // x = m @ W2 + b2 + t    (f32 for next LN + bf16 shadow for head GEMM)
    gemm_bf16<0, true, 3><<<dim3(D / 64, M / 128), blk, 0, stream>>>(
        bufMhb, W2T, b2 + (size_t)l * D, bufT, bufX, bufXb, M, D, DFF);
  }

  // logits = x @ W_out + b_out ; out = softmax(logits)
  gemm_bf16<0, false, 1><<<dim3(V / 64, M / 128), blk, 0, stream>>>(
      bufXb, WoutT, b_out, nullptr, bufL, nullptr, M, V, D);
  softmax_rows<<<M, 128, 0, stream>>>(bufL, out, V);
}
